// Attention_43593918054481
// MI455X (gfx1250) — matrix operand order repaired and checked by reading, not compiled, not run on hardware
//
#include <hip/hip_runtime.h>
#include <hip/hip_bf16.h>

// ---------------------------------------------------------------------------
// GQA attention layer for MI455X (gfx1250, wave32, WMMA).
//   x@wq/wk/wv -> RoPE -> causal flash attention -> @wo
// bf16 WMMA (v_wmma_f32_16x16x32_bf16) with fp32 accumulation everywhere.
// All LDS staging is contiguous b128 traffic via GLOBAL_LOAD_ASYNC_TO_LDS_B128
// (ASYNCcnt-tracked), double-buffered in the GEMM with a peeled epilogue so
// the steady-state loop is branch-free.
// ---------------------------------------------------------------------------

#define DIM   2048
#define SEQ   2048
#define NH    32
#define NKV   8
#define HD    64
#define KVDIM (NKV * HD)   // 512

typedef __bf16 bf16_t;
typedef __attribute__((ext_vector_type(16))) __bf16 v16bf;
typedef __attribute__((ext_vector_type(8)))  __bf16 v8bf;
typedef __attribute__((ext_vector_type(8)))  float  v8f;
typedef int vint4_t __attribute__((vector_size(16)));   // matches builtin param

#if defined(__has_builtin)
#  if __has_builtin(__builtin_amdgcn_global_load_async_to_lds_b128)
#    define HAVE_ASYNC_LDS 1
#  endif
#  if __has_builtin(__builtin_amdgcn_s_wait_asynccnt)
#    define HAVE_WAIT_ASYNC_BUILTIN 1
#  endif
#endif

static __device__ inline void async_copy16(bf16_t* lds, const bf16_t* gsrc) {
#if defined(HAVE_ASYNC_LDS)
  __builtin_amdgcn_global_load_async_to_lds_b128(
      (__attribute__((address_space(1))) vint4_t*)gsrc,
      (__attribute__((address_space(3))) vint4_t*)lds,
      /*offset=*/0, /*cpol=*/0);
#else
  *reinterpret_cast<uint4*>(lds) = *reinterpret_cast<const uint4*>(gsrc);
#endif
}

template <int N>
static __device__ inline void wait_asynccnt_le() {
#if defined(HAVE_ASYNC_LDS)
#  if defined(HAVE_WAIT_ASYNC_BUILTIN)
  __builtin_amdgcn_s_wait_asynccnt((short)N);
#  else
  asm volatile("s_wait_asynccnt %0" ::"i"(N) : "memory");
#  endif
#endif
}

static __device__ inline v8f v8f_zero() {
  v8f z;
#pragma unroll
  for (int i = 0; i < 8; ++i) z[i] = 0.0f;
  return z;
}

static __device__ inline v8bf load8(const bf16_t* p) {
  return *reinterpret_cast<const v8bf*>(p);
}

static __device__ inline v16bf cat16(v8bf lo, v8bf hi) {
  return __builtin_shufflevector(lo, hi, 0, 1, 2, 3, 4, 5, 6, 7,
                                 8, 9, 10, 11, 12, 13, 14, 15);
}

static __device__ inline v8f wmma_bf16(v16bf a, v16bf b, v8f c) {
  return __builtin_amdgcn_wmma_f32_16x16x32_bf16(
      false, a, false, b, (short)0, c, false, false);
}

// ---------------------------------------------------------------------------
// fp32 -> bf16 elementwise conversion (activations, row-major kept)
// ---------------------------------------------------------------------------
__global__ __launch_bounds__(256) void cvt_bf16_kernel(const float* __restrict__ in,
                                                       bf16_t* __restrict__ out,
                                                       int n) {
  int idx = blockIdx.x * 256 + threadIdx.x;
  if (idx < n) out[idx] = (bf16_t)in[idx];
}

// ---------------------------------------------------------------------------
// fp32 [R][C] -> bf16 [C][R] tiled transpose+convert (weights, done once).
// ---------------------------------------------------------------------------
__global__ __launch_bounds__(256) void cvt_t_bf16_kernel(const float* __restrict__ in,
                                                         bf16_t* __restrict__ out,
                                                         int R, int C) {
  __shared__ bf16_t tile[32][33];
  const int c0 = blockIdx.x * 32;
  const int r0 = blockIdx.y * 32;
  const int tx = threadIdx.x & 31;
  const int ty = threadIdx.x >> 5;   // 0..7
#pragma unroll
  for (int i = 0; i < 32; i += 8)
    tile[ty + i][tx] = (bf16_t)in[(size_t)(r0 + ty + i) * C + c0 + tx];
  __syncthreads();
#pragma unroll
  for (int i = 0; i < 32; i += 8)
    out[(size_t)(c0 + ty + i) * R + r0 + tx] = tile[tx][ty + i];
}

// ---------------------------------------------------------------------------
// RoPE (fp32 in, bf16 out); attention scale folded into Q.
// ---------------------------------------------------------------------------
__global__ __launch_bounds__(256) void rope_kernel(const float* __restrict__ in,
                                                   bf16_t* __restrict__ out,
                                                   const float* __restrict__ fcos,
                                                   const float* __restrict__ fsin,
                                                   int nheads, float scale) {
  int idx = blockIdx.x * 256 + threadIdx.x;
  int i = idx & 31;
  int h = (idx >> 5) % nheads;
  int s = idx / (32 * nheads);
  size_t base = (size_t)s * (size_t)(nheads * HD) + (size_t)h * HD + 2 * i;
  float c = fcos[s * 32 + i];
  float sn = fsin[s * 32 + i];
  float a = in[base];
  float b = in[base + 1];
  out[base]     = (bf16_t)((a * c - b * sn) * scale);
  out[base + 1] = (bf16_t)((a * sn + b * c) * scale);
}

// ---------------------------------------------------------------------------
// bf16 GEMM: C[M,N] = A[M,K] @ Bt[N,K]^T (B supplied pre-transposed).
// 256 threads (8 waves), 128x128 tile, K-step 32, double-buffered async LDS,
// last K-iteration peeled so the main loop is branch-free.
// ct != 0: store bf16 C transposed as C[n][m] (used to produce V^T).
// ---------------------------------------------------------------------------
__global__ __launch_bounds__(256) void gemm_bf16_kernel(const bf16_t* __restrict__ A,
                                                        const bf16_t* __restrict__ Bt,
                                                        float* __restrict__ Cf,
                                                        bf16_t* __restrict__ Cb,
                                                        int M, int N, int K, int ct) {
  __shared__ bf16_t As[2][128][32];   // [buf][m][k]
  __shared__ bf16_t Bs[2][128][32];   // [buf][n][k]

  const int tid  = threadIdx.x;
  const int lane = tid & 31;
  const int wave = tid >> 5;
  const int wm   = wave & 3;
  const int wn   = wave >> 2;
  const int half = lane >> 4;
  const int ln   = lane & 15;

  const int bm = blockIdx.y * 128;
  const int bn = blockIdx.x * 128;

  v8f acc[2][4];
#pragma unroll
  for (int mi = 0; mi < 2; ++mi)
#pragma unroll
    for (int ni = 0; ni < 4; ++ni) acc[mi][ni] = v8f_zero();

  // staging: row = tid>>1 (0..127), col = (tid&1)*16 : 2x16B per thread/tile
  const int srow = tid >> 1, scol = (tid & 1) * 16;
  const bf16_t* aP = A  + (size_t)(bm + srow) * K + scol;
  const bf16_t* bP = Bt + (size_t)(bn + srow) * K + scol;

  auto stage = [&](int buf) {
    async_copy16(&As[buf][srow][scol],     aP);
    async_copy16(&As[buf][srow][scol + 8], aP + 8);
    async_copy16(&Bs[buf][srow][scol],     bP);
    async_copy16(&Bs[buf][srow][scol + 8], bP + 8);
  };

  auto compute = [&](int buf) {
    v16bf afrag[2], bfrag[4];
#pragma unroll
    for (int mi = 0; mi < 2; ++mi) {
      const bf16_t* p = &As[buf][wm * 32 + mi * 16 + ln][0];
      afrag[mi] = cat16(load8(p + half * 8), load8(p + 16 + half * 8));
    }
#pragma unroll
    for (int ni = 0; ni < 4; ++ni) {
      const bf16_t* p = &Bs[buf][wn * 64 + ni * 16 + ln][0];
      bfrag[ni] = cat16(load8(p + half * 8), load8(p + 16 + half * 8));
    }
#pragma unroll
    for (int mi = 0; mi < 2; ++mi)
#pragma unroll
      for (int ni = 0; ni < 4; ++ni)
        acc[mi][ni] = wmma_bf16(afrag[mi], bfrag[ni], acc[mi][ni]);
  };

  const int nk = K >> 5;
  stage(0);
  // steady state: branch-free (stage next -> wait<=4 -> barrier -> WMMA -> barrier)
  for (int t = 0; t < nk - 1; ++t) {
    aP += 32;
    bP += 32;
    stage((t + 1) & 1);
    wait_asynccnt_le<4>();   // tile t resident (in-order), tile t+1 in flight
    __syncthreads();
    compute(t & 1);
    __syncthreads();         // protect buffer before next overwrite
  }
  // peeled last tile
  wait_asynccnt_le<0>();
  __syncthreads();
  compute((nk - 1) & 1);

#pragma unroll
  for (int mi = 0; mi < 2; ++mi) {
#pragma unroll
    for (int ni = 0; ni < 4; ++ni) {
#pragma unroll
      for (int r = 0; r < 8; ++r) {
        int row = bm + wm * 32 + mi * 16 + r + 8 * half;
        int col = bn + wn * 64 + ni * 16 + ln;
        float v = acc[mi][ni][r];
        if (Cb) {
          if (ct) Cb[(size_t)col * M + row] = (bf16_t)v;
          else    Cb[(size_t)row * N + col] = (bf16_t)v;
        } else {
          Cf[(size_t)row * N + col] = v;
        }
      }
    }
  }
}

// ---------------------------------------------------------------------------
// Causal flash attention (GQA). Grid (S/64, NH), 128 threads (4 waves).
// Wave: 16 Q rows x head_dim 64. Q frags register-resident; K and V^T tiles
// staged to LDS with contiguous async b128 copies.
//   qb: [S][NH*HD]  kb: [S][NKV*HD]  vt: [NKV*HD][S]  ob: [S][NH*HD]
// ---------------------------------------------------------------------------
__global__ __launch_bounds__(128) void attn_kernel(const bf16_t* __restrict__ qb,
                                                   const bf16_t* __restrict__ kb,
                                                   const bf16_t* __restrict__ vt,
                                                   bf16_t* __restrict__ ob) {
  __shared__ bf16_t Ks[64][64];     // [kv][d]
  __shared__ bf16_t Vt[64][64];     // [d][kv]
  __shared__ bf16_t Ps[4][16][64];  // per-wave P tile (C->A relayout)

  const int tid  = threadIdx.x;
  const int lane = tid & 31;
  const int wave = tid >> 5;
  const int half = lane >> 4;
  const int ln   = lane & 15;

  const int qh    = blockIdx.y;
  const int kvh   = qh >> 2;
  const int qblk  = blockIdx.x;
  const int qbase = qblk * 64 + wave * 16;

  v16bf qf[2];
  {
    const bf16_t* qrow = qb + (size_t)(qbase + ln) * DIM + qh * HD;
#pragma unroll
    for (int ks = 0; ks < 2; ++ks)
      qf[ks] = cat16(load8(qrow + ks * 32 + half * 8),
                     load8(qrow + ks * 32 + 16 + half * 8));
  }

  float m_r[8], l_r[8];
#pragma unroll
  for (int r = 0; r < 8; ++r) { m_r[r] = -1e30f; l_r[r] = 0.0f; }
  v8f oacc[4];
#pragma unroll
  for (int ni = 0; ni < 4; ++ni) oacc[ni] = v8f_zero();

  // staging coords: 128 threads, 64 rows x 64 cols, 64B (4x16B) per thread
  const int srow = tid >> 1, scb = (tid & 1) * 32;
  const bf16_t* kSrc = kb + (size_t)srow * KVDIM + kvh * HD + scb;   // [kv][d]
  const bf16_t* vSrc = vt + (size_t)(kvh * HD + srow) * SEQ + scb;   // [d][s]

  const int nblk = qblk + 1;
  for (int b = 0; b < nblk; ++b) {
    const int kvbase = b * 64;
#pragma unroll
    for (int j = 0; j < 4; ++j)
      async_copy16(&Ks[srow][scb + j * 8], kSrc + (size_t)kvbase * KVDIM + j * 8);
#pragma unroll
    for (int j = 0; j < 4; ++j)
      async_copy16(&Vt[srow][scb + j * 8], vSrc + kvbase + j * 8);
    wait_asynccnt_le<0>();
    __syncthreads();

    // ---- S = Q @ K^T ----
    v8f sacc[4];
#pragma unroll
    for (int ni = 0; ni < 4; ++ni) sacc[ni] = v8f_zero();
#pragma unroll
    for (int ks = 0; ks < 2; ++ks) {
#pragma unroll
      for (int ni = 0; ni < 4; ++ni) {
        const bf16_t* p = &Ks[ni * 16 + ln][ks * 32];
        v16bf bk = cat16(load8(p + half * 8), load8(p + 16 + half * 8));
        sacc[ni] = wmma_bf16(qf[ks], bk, sacc[ni]);
      }
    }

    // ---- causal mask + online softmax ----
    float corr[8];
#pragma unroll
    for (int r = 0; r < 8; ++r) {
      int row = qbase + r + 8 * half;
      float mx = -1e30f;
#pragma unroll
      for (int ni = 0; ni < 4; ++ni) {
        int col = kvbase + ni * 16 + ln;
        float sv = sacc[ni][r];
        if (col > row) { sv = -1e30f; sacc[ni][r] = sv; }
        mx = fmaxf(mx, sv);
      }
      mx = fmaxf(mx, __shfl_xor(mx, 1, 32));
      mx = fmaxf(mx, __shfl_xor(mx, 2, 32));
      mx = fmaxf(mx, __shfl_xor(mx, 4, 32));
      mx = fmaxf(mx, __shfl_xor(mx, 8, 32));
      float mn = fmaxf(m_r[r], mx);
      corr[r] = __expf(m_r[r] - mn);
      float rs = 0.0f;
#pragma unroll
      for (int ni = 0; ni < 4; ++ni) {
        float p = __expf(sacc[ni][r] - mn);
        sacc[ni][r] = p;
        rs += p;
      }
      rs += __shfl_xor(rs, 1, 32);
      rs += __shfl_xor(rs, 2, 32);
      rs += __shfl_xor(rs, 4, 32);
      rs += __shfl_xor(rs, 8, 32);
      l_r[r] = l_r[r] * corr[r] + rs;
      m_r[r] = mn;
    }
#pragma unroll
    for (int ni = 0; ni < 4; ++ni)
#pragma unroll
      for (int r = 0; r < 8; ++r) oacc[ni][r] *= corr[r];

    // ---- P: C-layout -> A-layout via per-wave LDS tile ----
#pragma unroll
    for (int ni = 0; ni < 4; ++ni)
#pragma unroll
      for (int r = 0; r < 8; ++r)
        Ps[wave][r + 8 * half][ni * 16 + ln] = (bf16_t)sacc[ni][r];

    // ---- O += P @ V ----
#pragma unroll
    for (int ks = 0; ks < 2; ++ks) {
      const bf16_t* pp = &Ps[wave][ln][ks * 32];
      v16bf ap = cat16(load8(pp + half * 8), load8(pp + 16 + half * 8));
#pragma unroll
      for (int ni = 0; ni < 4; ++ni) {
        const bf16_t* vp = &Vt[ni * 16 + ln][ks * 32];
        v16bf bv = cat16(load8(vp + half * 8), load8(vp + 16 + half * 8));
        oacc[ni] = wmma_bf16(ap, bv, oacc[ni]);
      }
    }
    __syncthreads();
  }

#pragma unroll
  for (int ni = 0; ni < 4; ++ni) {
#pragma unroll
    for (int r = 0; r < 8; ++r) {
      int row = qbase + r + 8 * half;
      int col = qh * HD + ni * 16 + ln;
      ob[(size_t)row * DIM + col] = (bf16_t)(oacc[ni][r] / l_r[r]);
    }
  }
}

// ---------------------------------------------------------------------------
// Host-side orchestration
// ---------------------------------------------------------------------------
extern "C" void kernel_launch(void* const* d_in, const int* in_sizes, int n_in,
                              void* d_out, int out_size, void* d_ws, size_t ws_size,
                              hipStream_t stream) {
  const float* x  = (const float*)d_in[0];
  const float* wq = (const float*)d_in[1];
  const float* wk = (const float*)d_in[2];
  const float* wv = (const float*)d_in[3];
  const float* wo = (const float*)d_in[4];
  const float* fc = (const float*)d_in[5];
  const float* fs = (const float*)d_in[6];
  // d_in[7] = mask (causal handled analytically), d_in[8] = start_pos (== 0)

  char* ws = (char*)d_ws;
  size_t off = 0;
  auto alloc = [&](size_t bytes) -> void* {
    void* p = ws + off;
    off = (off + bytes + 255) & ~(size_t)255;
    return p;
  };

  bf16_t* xb  = (bf16_t*)alloc((size_t)SEQ * DIM * 2);
  bf16_t* wqt = (bf16_t*)alloc((size_t)DIM * DIM * 2);     // [N][K]
  bf16_t* wkt = (bf16_t*)alloc((size_t)KVDIM * DIM * 2);   // [N][K]
  bf16_t* wvt = (bf16_t*)alloc((size_t)KVDIM * DIM * 2);   // [N][K]
  bf16_t* wot = (bf16_t*)alloc((size_t)DIM * DIM * 2);     // [N][K]
  float*  qf  = (float*)alloc((size_t)SEQ * DIM * 4);
  float*  kf  = (float*)alloc((size_t)SEQ * KVDIM * 4);
  bf16_t* qbf = (bf16_t*)alloc((size_t)SEQ * DIM * 2);
  bf16_t* kbf = (bf16_t*)alloc((size_t)SEQ * KVDIM * 2);
  bf16_t* vtb = (bf16_t*)alloc((size_t)KVDIM * SEQ * 2);   // V^T [d][s]
  bf16_t* obf = (bf16_t*)alloc((size_t)SEQ * DIM * 2);
  (void)ws_size;

  // 1) downconvert: x row-major; weights transposed to [N][K]
  cvt_bf16_kernel<<<(SEQ * DIM) / 256, 256, 0, stream>>>(x, xb, SEQ * DIM);
  cvt_t_bf16_kernel<<<dim3(DIM / 32, DIM / 32), 256, 0, stream>>>(wq, wqt, DIM, DIM);
  cvt_t_bf16_kernel<<<dim3(KVDIM / 32, DIM / 32), 256, 0, stream>>>(wk, wkt, DIM, KVDIM);
  cvt_t_bf16_kernel<<<dim3(KVDIM / 32, DIM / 32), 256, 0, stream>>>(wv, wvt, DIM, KVDIM);
  cvt_t_bf16_kernel<<<dim3(DIM / 32, DIM / 32), 256, 0, stream>>>(wo, wot, DIM, DIM);

  // 2) QKV projections. Q,K -> fp32 (rope follows); V -> bf16 transposed.
  gemm_bf16_kernel<<<dim3(DIM / 128, SEQ / 128), 256, 0, stream>>>(
      xb, wqt, qf, nullptr, SEQ, DIM, DIM, 0);
  gemm_bf16_kernel<<<dim3(KVDIM / 128, SEQ / 128), 256, 0, stream>>>(
      xb, wkt, kf, nullptr, SEQ, KVDIM, DIM, 0);
  gemm_bf16_kernel<<<dim3(KVDIM / 128, SEQ / 128), 256, 0, stream>>>(
      xb, wvt, nullptr, vtb, SEQ, KVDIM, DIM, 1);

  // 3) RoPE (fold 1/sqrt(HD)=0.125 into Q)
  rope_kernel<<<(SEQ * NH * 32) / 256, 256, 0, stream>>>(qf, qbf, fc, fs, NH, 0.125f);
  rope_kernel<<<(SEQ * NKV * 32) / 256, 256, 0, stream>>>(kf, kbf, fc, fs, NKV, 1.0f);

  // 4) causal flash attention
  attn_kernel<<<dim3(SEQ / 64, NH), 128, 0, stream>>>(qbf, kbf, vtb, obf);

  // 5) output projection -> fp32 d_out
  gemm_bf16_kernel<<<dim3(DIM / 128, SEQ / 128), 256, 0, stream>>>(
      obf, wot, (float*)d_out, nullptr, SEQ, DIM, DIM, 0);
}
